// KNNAttentionAggBeforeMLP_28458453303527
// MI455X (gfx1250) — matrix operand order repaired and checked by reading, not compiled, not run on hardware
//
#include <hip/hip_runtime.h>
#include <hip/hip_bf16.h>
#include <math.h>

typedef __attribute__((ext_vector_type(16))) _Float16 v16h;
typedef __attribute__((ext_vector_type(8)))  _Float16 v8h;
typedef __attribute__((ext_vector_type(8)))  float    v8f;
typedef _Float16 h16;

#define BB     2
#define SS     1024
#define DMODEL 1024
#define NH     16
#define HDIM   64
#define MEMM   32
#define EPSL   1e-5f

// ---------------------------------------------------------------------------
// WMMA fragment helpers (layouts per CDNA5 ISA §7.12.2, wave32)
// A: 16x32 f16.  lane L: row = L&15, hi = L>>4; element e -> K = (e/8)*16 + hi*8 + e%8
//    => two contiguous 8-half (16B) loads at row*lda + hi*8 and +16.
// B: 32x16 f16 stored as Bt[N][K]. lane L: col = L&15; element e -> K = (e/8)*16 + hi*8 + e%8
//    => two contiguous 8-half (16B) loads at col*ldb + hi*8 and +16.
// C/D: 16x16 f32, v8f: element r -> row = r + 8*hi, col = L&15.
// ---------------------------------------------------------------------------
__device__ __forceinline__ v16h load_a_frag(const h16* a, int lda) {
  int lane = threadIdx.x & 31;
  int row  = lane & 15, hi = lane >> 4;
  const h16* p = a + (size_t)row * lda + hi * 8;
  v8h x0 = *(const v8h*)p;
  v8h x1 = *(const v8h*)(p + 16);
  v16h r;
#pragma unroll
  for (int i = 0; i < 8; ++i) { r[i] = x0[i]; r[8 + i] = x1[i]; }
  return r;
}

__device__ __forceinline__ v16h load_b_frag(const h16* bt, int ldb) {
  int lane = threadIdx.x & 31;
  int col  = lane & 15, hi = lane >> 4;
  const h16* p = bt + (size_t)col * ldb + hi * 8;
  v8h x0 = *(const v8h*)p;
  v8h x1 = *(const v8h*)(p + 16);
  v16h r;
#pragma unroll
  for (int i = 0; i < 8; ++i) { r[i] = x0[i]; r[8 + i] = x1[i]; }
  return r;
}

__device__ __forceinline__ v8f wmma16(v16h a, v16h b, v8f c) {
  return __builtin_amdgcn_wmma_f32_16x16x32_f16(false, a, false, b, (short)0, c,
                                                false, false);
}

// ---------------------------------------------------------------------------
// Weight transpose + fp32 -> fp16:  in[K][N] -> out[N][K]
// ---------------------------------------------------------------------------
__global__ void k_transpose_h16(const float* __restrict__ in, h16* __restrict__ out,
                                int K, int N) {
  size_t idx = (size_t)blockIdx.x * blockDim.x + threadIdx.x;
  if (idx >= (size_t)K * N) return;
  int k = (int)(idx / N), n = (int)(idx % N);
  out[(size_t)n * K + k] = (h16)in[idx];
}

// ---------------------------------------------------------------------------
// LayerNorm rows (fp32 in -> fp16 out), one block per row
// ---------------------------------------------------------------------------
__global__ void k_ln_rows(const float* __restrict__ in, const float* __restrict__ gam,
                          const float* __restrict__ bet, h16* __restrict__ out,
                          int cols) {
  int row = blockIdx.x;
  const float* x = in + (size_t)row * cols;
  __shared__ float s1[256], s2[256];
  int tid = threadIdx.x;
  float a = 0.f, b = 0.f;
  for (int c = tid; c < cols; c += 256) { float v = x[c]; a += v; b += v * v; }
  s1[tid] = a; s2[tid] = b; __syncthreads();
  for (int st = 128; st > 0; st >>= 1) {
    if (tid < st) { s1[tid] += s1[tid + st]; s2[tid] += s2[tid + st]; }
    __syncthreads();
  }
  float mean = s1[0] / cols;
  float var  = s2[0] / cols - mean * mean;
  float rs   = rsqrtf(var + EPSL);
  for (int c = tid; c < cols; c += 256)
    out[(size_t)row * cols + c] = (h16)(((x[c] - mean) * rs) * gam[c] + bet[c]);
}

// ---------------------------------------------------------------------------
// Generic WMMA GEMM: C[M][N] = A[M][K](f16) * Bt[N][K](f16) + bias, with epilogue.
// One wave computes a 32x64 output tile (2 A-frags x 4 B-frags = 8 WMMAs/k-step;
// B-fragments reused across both row tiles -> ~21 FLOP per fragment byte).
// mode: 0=store f32, 1=gate-combine, 2=gelu->f16, 3=+residual->f32
// ---------------------------------------------------------------------------
__global__ void k_gemm_wmma(const h16* __restrict__ A, const h16* __restrict__ Bt,
                            const float* __restrict__ bias,
                            float* __restrict__ outf, h16* __restrict__ outh,
                            const float* __restrict__ res, const float* __restrict__ mem,
                            const float* __restrict__ gptr,
                            int Mrows, int Ncols, int Kdim, int mode) {
  int wave   = (int)((blockIdx.x * blockDim.x + threadIdx.x) >> 5);
  int nWaves = (int)((gridDim.x * blockDim.x) >> 5);
  int nTiles = Ncols >> 6;
  int total  = (Mrows >> 5) * nTiles;
  int lane = threadIdx.x & 31;
  int col  = lane & 15, hi = lane >> 4;
  float g = (gptr != nullptr) ? gptr[0] : 0.f;

  for (int t = wave; t < total; t += nWaves) {
    int mt = t / nTiles, nt = t % nTiles;
    int m0 = mt << 5, n0 = nt << 6;
    v8f zero = {};
    v8f acc[2][4];
#pragma unroll
    for (int mi = 0; mi < 2; ++mi)
#pragma unroll
      for (int n = 0; n < 4; ++n) acc[mi][n] = zero;

    const h16* Ab0 = A + (size_t)m0 * Kdim;
    const h16* Ab1 = A + (size_t)(m0 + 16) * Kdim;
    for (int k = 0; k < Kdim; k += 32) {
      v16h a0 = load_a_frag(Ab0 + k, Kdim);
      v16h a1 = load_a_frag(Ab1 + k, Kdim);
#pragma unroll
      for (int n = 0; n < 4; ++n) {
        v16h b = load_b_frag(Bt + (size_t)(n0 + n * 16) * Kdim + k, Kdim);
        acc[0][n] = wmma16(a0, b, acc[0][n]);
        acc[1][n] = wmma16(a1, b, acc[1][n]);
      }
    }
#pragma unroll
    for (int mi = 0; mi < 2; ++mi) {
#pragma unroll
      for (int n = 0; n < 4; ++n) {
        int nIdx = n0 + n * 16 + col;
        float bv = bias ? bias[nIdx] : 0.f;
#pragma unroll
        for (int r = 0; r < 8; ++r) {
          int mIdx = m0 + mi * 16 + r + 8 * hi;
          size_t o = (size_t)mIdx * Ncols + nIdx;
          float v = acc[mi][n][r] + bv;
          if (mode == 0) {
            outf[o] = v;
          } else if (mode == 1) {
            outf[o] = res[o] + (1.f - g) * v + g * mem[o];
          } else if (mode == 2) {
            float t3 = tanhf(0.7978845608028654f * (v + 0.044715f * v * v * v));
            outh[o] = (h16)(0.5f * v * (1.f + t3));
          } else {
            outf[o] = res[o] + v;
          }
        }
      }
    }
  }
}

// ---------------------------------------------------------------------------
// Repack qkv fp32 -> per-head fp16: q (scaled by 1/8) [B,H,S,64], k [B,H,S,64],
// v transposed [B,H,64,S]
// ---------------------------------------------------------------------------
__global__ void k_repack_qkv(const float* __restrict__ qkv, h16* __restrict__ q16,
                             h16* __restrict__ k16, h16* __restrict__ vT) {
  size_t idx = (size_t)blockIdx.x * blockDim.x + threadIdx.x;
  if (idx >= (size_t)BB * SS * DMODEL) return;
  int c = (int)(idx % DMODEL);
  size_t bs = idx / DMODEL;
  int s = (int)(bs % SS); int b = (int)(bs / SS);
  int h = c / HDIM, d = c % HDIM;
  size_t base = bs * (size_t)(3 * DMODEL);
  size_t bh = (size_t)b * NH + h;
  q16[(bh * SS + s) * HDIM + d] = (h16)(qkv[base + c] * 0.125f);
  k16[(bh * SS + s) * HDIM + d] = (h16)qkv[base + DMODEL + c];
  vT [(bh * HDIM + d) * SS + s] = (h16)qkv[base + 2 * DMODEL + c];
}

// ---------------------------------------------------------------------------
// Causal flash attention with WMMA. One wave: 16-query tile of one (b,h).
// Output fp16, merged layout [B,S,DM].
// ---------------------------------------------------------------------------
__global__ void k_flash_attn(const h16* __restrict__ q16, const h16* __restrict__ k16,
                             const h16* __restrict__ vT, h16* __restrict__ std16) {
  __shared__ __align__(16) h16 plds[8][16 * 32];
  int wv  = threadIdx.x >> 5;
  int wid = blockIdx.x * 8 + wv;          // 0 .. BB*NH*(SS/16)-1
  int bh  = wid >> 6;                     // SS/16 == 64 tiles per (b,h)
  int qt  = wid & 63;
  int lane = threadIdx.x & 31;
  int col = lane & 15, hi = lane >> 4;

  const h16* Q = q16 + (size_t)bh * SS * HDIM;
  const h16* K = k16 + (size_t)bh * SS * HDIM;
  const h16* V = vT  + (size_t)bh * HDIM * SS;
  int q0 = qt << 4;

  v16h aq0 = load_a_frag(Q + (size_t)q0 * HDIM, HDIM);
  v16h aq1 = load_a_frag(Q + (size_t)q0 * HDIM + 32, HDIM);

  v8f zero = {};
  v8f O[4]; O[0] = zero; O[1] = zero; O[2] = zero; O[3] = zero;
  float mi[8], li[8];
#pragma unroll
  for (int r = 0; r < 8; ++r) { mi[r] = -INFINITY; li[r] = 0.f; }
  h16* P = &plds[wv][0];

  for (int kb = 0; kb <= q0 + 15; kb += 32) {
    v8f s0 = zero, s1 = zero;
    v16h b;
    b = load_b_frag(K + (size_t)kb * HDIM, HDIM);        s0 = wmma16(aq0, b, s0);
    b = load_b_frag(K + (size_t)kb * HDIM + 32, HDIM);   s0 = wmma16(aq1, b, s0);
    b = load_b_frag(K + (size_t)(kb + 16) * HDIM, HDIM);      s1 = wmma16(aq0, b, s1);
    b = load_b_frag(K + (size_t)(kb + 16) * HDIM + 32, HDIM); s1 = wmma16(aq1, b, s1);

    int key0 = kb + col, key1 = kb + 16 + col;
#pragma unroll
    for (int r = 0; r < 8; ++r) {
      int qm = q0 + r + 8 * hi;
      float v0 = (key0 <= qm) ? s0[r] : -INFINITY;
      float v1 = (key1 <= qm) ? s1[r] : -INFINITY;
      float mx = fmaxf(v0, v1);
      mx = fmaxf(mx, __shfl_xor(mx, 1));
      mx = fmaxf(mx, __shfl_xor(mx, 2));
      mx = fmaxf(mx, __shfl_xor(mx, 4));
      mx = fmaxf(mx, __shfl_xor(mx, 8));
      float mnew = fmaxf(mi[r], mx);
      float sc = __expf(mi[r] - mnew);
      float p0 = __expf(v0 - mnew);
      float p1 = __expf(v1 - mnew);
      float rs = p0 + p1;
      rs += __shfl_xor(rs, 1);
      rs += __shfl_xor(rs, 2);
      rs += __shfl_xor(rs, 4);
      rs += __shfl_xor(rs, 8);
      li[r] = li[r] * sc + rs;
      mi[r] = mnew;
#pragma unroll
      for (int n = 0; n < 4; ++n) O[n][r] = O[n][r] * sc;
      int mloc = r + 8 * hi;
      P[mloc * 32 + col]      = (h16)p0;
      P[mloc * 32 + 16 + col] = (h16)p1;
    }
    v16h aP = load_a_frag(P, 32);
#pragma unroll
    for (int n = 0; n < 4; ++n) {
      v16h bv = load_b_frag(V + (size_t)(n * 16) * SS + kb, SS);
      O[n] = wmma16(aP, bv, O[n]);
    }
  }

  int b_ = bh >> 4, h_ = bh & 15;
#pragma unroll
  for (int n = 0; n < 4; ++n) {
    int d = n * 16 + col;
#pragma unroll
    for (int r = 0; r < 8; ++r) {
      int qm = q0 + r + 8 * hi;
      float val = O[n][r] / li[r];
      std16[((size_t)(b_ * SS + qm)) * DMODEL + h_ * HDIM + d] = (h16)val;
    }
  }
}

// ---------------------------------------------------------------------------
// KNN-memory attention (bandwidth bound on 536MB mem_kv): one wave per (b,s,h),
// lane = candidate m for scores, lane = d / d+32 for value accumulation.
// ---------------------------------------------------------------------------
__global__ void k_mem_attn(const float* __restrict__ qkv, const float* __restrict__ memkv,
                           float* __restrict__ out) {
  int wv = threadIdx.x >> 5;
  int wid = blockIdx.x * 8 + wv;           // 0 .. BB*SS*NH-1
  int lane = threadIdx.x & 31;
  int h = wid % NH;
  size_t bs = (size_t)(wid / NH);
  const float* q    = qkv + bs * (size_t)(3 * DMODEL) + h * HDIM;
  const float* krow = memkv + ((bs * MEMM + lane) * 2) * (size_t)DMODEL + h * HDIM;
  float dot = 0.f;
#pragma unroll
  for (int d = 0; d < HDIM; d += 4) {
    float4 qv = *(const float4*)(q + d);
    float4 kv = *(const float4*)(krow + d);
    dot += qv.x * kv.x + qv.y * kv.y + qv.z * kv.z + qv.w * kv.w;
  }
  dot *= 0.125f;
  float mx = dot;
  mx = fmaxf(mx, __shfl_xor(mx, 1));
  mx = fmaxf(mx, __shfl_xor(mx, 2));
  mx = fmaxf(mx, __shfl_xor(mx, 4));
  mx = fmaxf(mx, __shfl_xor(mx, 8));
  mx = fmaxf(mx, __shfl_xor(mx, 16));
  float e = __expf(dot - mx);
  float s = e;
  s += __shfl_xor(s, 1);
  s += __shfl_xor(s, 2);
  s += __shfl_xor(s, 4);
  s += __shfl_xor(s, 8);
  s += __shfl_xor(s, 16);
  float w = e / s;

  float acc0 = 0.f, acc1 = 0.f;
#pragma unroll 4
  for (int m = 0; m < MEMM; ++m) {
    float wm = __shfl(w, m);
    const float* vrow = memkv + ((bs * MEMM + m) * 2 + 1) * (size_t)DMODEL + h * HDIM;
    acc0 += wm * vrow[lane];
    acc1 += wm * vrow[lane + 32];
  }
  out[bs * DMODEL + h * HDIM + lane]      = acc0;
  out[bs * DMODEL + h * HDIM + lane + 32] = acc1;
}

// ---------------------------------------------------------------------------
extern "C" void kernel_launch(void* const* d_in, const int* in_sizes, int n_in,
                              void* d_out, int out_size, void* d_ws, size_t ws_size,
                              hipStream_t stream) {
  const float* x       = (const float*)d_in[0];
  const float* memkv   = (const float*)d_in[1];
  const float* gval    = (const float*)d_in[2];
  const float* ln1g    = (const float*)d_in[3];
  const float* ln1b    = (const float*)d_in[4];
  const float* cattn_w = (const float*)d_in[5];
  const float* cattn_b = (const float*)d_in[6];
  const float* cproj_w = (const float*)d_in[7];
  const float* cproj_b = (const float*)d_in[8];
  const float* ln2g    = (const float*)d_in[9];
  const float* ln2b    = (const float*)d_in[10];
  const float* fc_w    = (const float*)d_in[11];
  const float* fc_b    = (const float*)d_in[12];
  const float* proj_w  = (const float*)d_in[13];
  const float* proj_b  = (const float*)d_in[14];
  float* out = (float*)d_out;

  char* wsp = (char*)d_ws;
  auto carve = [&](size_t bytes) -> char* {
    char* p = wsp; wsp += (bytes + 255) & ~(size_t)255; return p;
  };
  const int ROWS = BB * SS;                 // 2048
  h16*   wT_attn  = (h16*)carve((size_t)3 * DMODEL * DMODEL * 2);
  h16*   wT_cproj = (h16*)carve((size_t)DMODEL * DMODEL * 2);
  h16*   wT_fc    = (h16*)carve((size_t)4 * DMODEL * DMODEL * 2);
  h16*   wT_proj  = (h16*)carve((size_t)4 * DMODEL * DMODEL * 2);
  h16*   hln      = (h16*)carve((size_t)ROWS * DMODEL * 2);
  float* qkv      = (float*)carve((size_t)ROWS * 3 * DMODEL * 4);
  h16*   q16      = (h16*)carve((size_t)ROWS * DMODEL * 2);
  h16*   k16      = (h16*)carve((size_t)ROWS * DMODEL * 2);
  h16*   vT16     = (h16*)carve((size_t)ROWS * DMODEL * 2);
  h16*   std16    = (h16*)carve((size_t)ROWS * DMODEL * 2);
  float* memo     = (float*)carve((size_t)ROWS * DMODEL * 4);
  float* h1       = (float*)carve((size_t)ROWS * DMODEL * 4);
  h16*   h2ln     = (h16*)carve((size_t)ROWS * DMODEL * 2);
  h16*   act      = (h16*)carve((size_t)ROWS * 4 * DMODEL * 2);

  // 1) weight transposes (fp32 -> fp16, [K][N] -> [N][K])
  k_transpose_h16<<<(DMODEL * 3 * DMODEL + 255) / 256, 256, 0, stream>>>(cattn_w, wT_attn, DMODEL, 3 * DMODEL);
  k_transpose_h16<<<(DMODEL * DMODEL + 255) / 256, 256, 0, stream>>>(cproj_w, wT_cproj, DMODEL, DMODEL);
  k_transpose_h16<<<(DMODEL * 4 * DMODEL + 255) / 256, 256, 0, stream>>>(fc_w, wT_fc, DMODEL, 4 * DMODEL);
  k_transpose_h16<<<(4 * DMODEL * DMODEL + 255) / 256, 256, 0, stream>>>(proj_w, wT_proj, 4 * DMODEL, DMODEL);

  // 2) LN1
  k_ln_rows<<<ROWS, 256, 0, stream>>>(x, ln1g, ln1b, hln, DMODEL);

  // 3) qkv GEMM: [2048,1024] x [1024,3072]
  {
    int tiles = (ROWS / 32) * (3 * DMODEL / 64);     // 64 * 48 = 3072
    k_gemm_wmma<<<tiles / 8, 256, 0, stream>>>(hln, wT_attn, cattn_b, qkv, nullptr,
                                               nullptr, nullptr, nullptr,
                                               ROWS, 3 * DMODEL, DMODEL, 0);
  }

  // 4) repack q/k/v into per-head fp16 (q pre-scaled by 1/sqrt(64))
  k_repack_qkv<<<(ROWS * DMODEL + 255) / 256, 256, 0, stream>>>(qkv, q16, k16, vT16);

  // 5) causal flash attention (WMMA) -> std16 fp16 merged
  k_flash_attn<<<(BB * NH * (SS / 16)) / 8, 256, 0, stream>>>(q16, k16, vT16, std16);

  // 6) KNN memory attention -> memo fp32 merged
  k_mem_attn<<<(BB * SS * NH) / 8, 256, 0, stream>>>(qkv, memkv, memo);

  // 7) c_proj GEMM + gated combine + residual -> h1
  {
    int tiles = (ROWS / 32) * (DMODEL / 64);         // 64 * 16 = 1024
    k_gemm_wmma<<<tiles / 8, 256, 0, stream>>>(std16, wT_cproj, cproj_b, h1, nullptr,
                                               x, memo, gval,
                                               ROWS, DMODEL, DMODEL, 1);
  }

  // 8) LN2
  k_ln_rows<<<ROWS, 256, 0, stream>>>(h1, ln2g, ln2b, h2ln, DMODEL);

  // 9) fc GEMM + gelu(tanh) -> act fp16
  {
    int tiles = (ROWS / 32) * (4 * DMODEL / 64);     // 64 * 64 = 4096
    k_gemm_wmma<<<tiles / 8, 256, 0, stream>>>(h2ln, wT_fc, fc_b, nullptr, act,
                                               nullptr, nullptr, nullptr,
                                               ROWS, 4 * DMODEL, DMODEL, 2);
  }

  // 10) proj GEMM + bias + residual2 -> d_out fp32
  {
    int tiles = (ROWS / 32) * (DMODEL / 64);         // 64 * 16 = 1024
    k_gemm_wmma<<<tiles / 8, 256, 0, stream>>>(act, wT_proj, proj_b, out, nullptr,
                                               h1, nullptr, nullptr,
                                               ROWS, DMODEL, 4 * DMODEL, 3);
  }
}
